// StepwiseSAEBank_6622839570550
// MI455X (gfx1250) — matrix operand order repaired and checked by reading, not compiled, not run on hardware
//
#include <hip/hip_runtime.h>
#include <hip/hip_bf16.h>

#define S_STEPS 7
#define DMODEL  768
#define DSAE    24576
#define LDSK    40   // padded bf16 row stride for a 32-wide K panel (80B: 16B-aligned, conflict-free)

typedef __attribute__((ext_vector_type(16))) __bf16 v16bf;
typedef __attribute__((ext_vector_type(8)))  __bf16 v8bf;
typedef __attribute__((ext_vector_type(8)))  float  v8f;

// ---------------------------------------------------------------------------
// Load a 128-row x 32-col f32 panel from global, convert to bf16, store to LDS.
// 256 threads: each thread moves 16 floats (4x float4) -> 2x 16B ds stores.
// ---------------------------------------------------------------------------
__device__ __forceinline__ void load_panel_bf16(__bf16 (*dst)[LDSK],
                                                const float* __restrict__ srcRow0,
                                                long rstride, int tid) {
  const int lr = tid >> 1;            // row 0..127 (2 threads per row)
  const int lc = (tid & 1) << 4;      // col 0 or 16
  const float* g = srcRow0 + (long)lr * rstride + lc;
  float4 f0 = *(const float4*)(g + 0);
  float4 f1 = *(const float4*)(g + 4);
  float4 f2 = *(const float4*)(g + 8);
  float4 f3 = *(const float4*)(g + 12);
  __builtin_prefetch(g + 32, 0, 3);   // hint next K panel into near cache (global_prefetch_b8)
  v8bf p0, p1;
  p0[0] = (__bf16)f0.x; p0[1] = (__bf16)f0.y; p0[2] = (__bf16)f0.z; p0[3] = (__bf16)f0.w;
  p0[4] = (__bf16)f1.x; p0[5] = (__bf16)f1.y; p0[6] = (__bf16)f1.z; p0[7] = (__bf16)f1.w;
  p1[0] = (__bf16)f2.x; p1[1] = (__bf16)f2.y; p1[2] = (__bf16)f2.z; p1[3] = (__bf16)f2.w;
  p1[4] = (__bf16)f3.x; p1[5] = (__bf16)f3.y; p1[6] = (__bf16)f3.z; p1[7] = (__bf16)f3.w;
  *(v8bf*)&dst[lr][lc]     = p0;
  *(v8bf*)&dst[lr][lc + 8] = p1;
}

// ---------------------------------------------------------------------------
// NT GEMM, bf16 inputs / f32 accum:  C[s][m][n] = sum_k A[s][m][k]*B[s][n][k] + bias[s][n]
// Both operands are K-contiguous (fits WMMA wave32 fragment maps directly).
// Block = 256 threads (8 waves), tile = 128(M) x 128(N), K panel = 32, double buffered.
// Each wave owns 16 M-rows and all 8 N-subtiles. All 8 B fragments are preloaded
// into distinct registers (one ds clause + one wait), then 8 back-to-back WMMAs.
// ---------------------------------------------------------------------------
__global__ __launch_bounds__(256, 2)
void sae_gemm_nt_bf16(const float* __restrict__ A,  long a_sstr, long a_bstr,
                      const float* __restrict__ Bm, long b_sstr, long b_rstr,
                      const float* __restrict__ bias, long bias_sstr,
                      float* __restrict__ C, long c_sstr, long c_bstr,
                      int Kdim) {
  __shared__ __bf16 Ash[2][128][LDSK];
  __shared__ __bf16 Bsh[2][128][LDSK];

  const int tid  = threadIdx.x;
  const int wave = tid >> 5;
  const int lane = tid & 31;
  const int half = lane >> 4;   // wave32 half: selects K sub-ranges / M sub-rows
  const int l16  = lane & 15;

  const int  s  = blockIdx.z;
  const long m0 = (long)blockIdx.y * 128;
  const long n0 = (long)blockIdx.x * 128;

  const float* Abase = A  + (long)s * a_sstr + m0 * a_bstr;
  const float* Bbase = Bm + (long)s * b_sstr + n0 * b_rstr;

  v8f acc[8] = {};

  const int nk = Kdim >> 5;
  load_panel_bf16(Ash[0], Abase, a_bstr, tid);
  load_panel_bf16(Bsh[0], Bbase, b_rstr, tid);
  __syncthreads();

  for (int t = 0; t < nk; ++t) {
    const int cur = t & 1;
    if (t + 1 < nk) {   // stage next panel into the other buffer
      load_panel_bf16(Ash[cur ^ 1], Abase + (long)(t + 1) * 32, a_bstr, tid);
      load_panel_bf16(Bsh[cur ^ 1], Bbase + (long)(t + 1) * 32, b_rstr, tid);
    }

    // A fragment: 16x32 bf16. lane<16: M=lane, K={0..7,16..23}; lane>=16: M=lane-16, K={8..15,24..31}
    const __bf16* ar = &Ash[cur][wave * 16 + l16][0];
    v8bf alo = *(const v8bf*)(ar + half * 8);
    v8bf ahi = *(const v8bf*)(ar + 16 + half * 8);
    v16bf afrag = __builtin_shufflevector(alo, ahi, 0,1,2,3,4,5,6,7,8,9,10,11,12,13,14,15);

    // Preload ALL 8 B fragments (distinct registers) so the ds loads form one
    // clause with a single dscnt wait, instead of a wait before every WMMA.
    v16bf bfrag[8];
#pragma unroll
    for (int ct = 0; ct < 8; ++ct) {
      // B fragment: 32x16 bf16. lane<16: N=lane, K={0..7,16..23}; lane>=16: N=lane-16, K={8..15,24..31}
      const __bf16* br = &Bsh[cur][ct * 16 + l16][0];
      v8bf blo = *(const v8bf*)(br + half * 8);
      v8bf bhi = *(const v8bf*)(br + 16 + half * 8);
      bfrag[ct] = __builtin_shufflevector(blo, bhi, 0,1,2,3,4,5,6,7,8,9,10,11,12,13,14,15);
    }

    // 8 back-to-back WMMAs on the matrix pipe.
#pragma unroll
    for (int ct = 0; ct < 8; ++ct)
      acc[ct] = __builtin_amdgcn_wmma_f32_16x16x32_bf16(false, afrag, false, bfrag[ct],
                                                        (short)0, acc[ct], false, false);
    __syncthreads();
  }

  // Epilogue: C/D layout -> lane<16: N=l16, M=vgpr j; lane>=16: N=l16, M=8+j
  float* Crow = C + (long)s * c_sstr;
#pragma unroll
  for (int ct = 0; ct < 8; ++ct) {
    const long n = n0 + ct * 16 + l16;
    const float bv = bias ? bias[(long)s * bias_sstr + n] : 0.f;
#pragma unroll
    for (int j = 0; j < 8; ++j) {
      const long m = m0 + wave * 16 + half * 8 + j;
      Crow[m * c_bstr + n] = acc[ct][j] + bv;
    }
  }
}

// ---------------------------------------------------------------------------
// In-place top-k selection per (b,s) row of z (length 24576):
// 4-level radix select on monotone-mapped float bits finds the exact K-th
// largest key; then z[f] = (key >= thr && v > 0) ? v : 0  (== scatter of relu'd
// top-k values into zeros). Row re-reads hit L2 (per-WG working set 96KB).
// ---------------------------------------------------------------------------
__device__ __forceinline__ unsigned fkey(float x) {
  unsigned u = __float_as_uint(x);
  return (u >> 31) ? ~u : (u | 0x80000000u);
}

__global__ __launch_bounds__(256)
void sae_topk_inplace(float* __restrict__ z, const int* __restrict__ kptr) {
  __shared__ unsigned hist[256];
  __shared__ unsigned sh_prefix;
  __shared__ int      sh_need;

  const int tid = threadIdx.x;
  float* row = z + (size_t)blockIdx.x * DSAE;
  const int kval = kptr[0];

  unsigned prefix = 0;
  int need = kval;

  for (int lvl = 0; lvl < 4; ++lvl) {
    const int shift = 24 - 8 * lvl;
    const unsigned mk = (lvl == 0) ? 0u : (0xFFFFFFFFu << (32 - 8 * lvl));

    for (int i = tid; i < 256; i += 256) hist[i] = 0;
    __syncthreads();

    for (int i = tid; i < DSAE; i += 256) {
      const unsigned key = fkey(row[i]);
      if ((key & mk) == (prefix & mk))
        atomicAdd(&hist[(key >> shift) & 255u], 1u);
    }
    __syncthreads();

    if (tid == 0) {
      int run = 0, sel = 0, below = 0;
      for (int b = 255; b >= 0; --b) {
        const int c = (int)hist[b];
        if (run + c >= need || b == 0) { sel = b; below = run; break; }
        run += c;
      }
      sh_prefix = prefix | ((unsigned)sel << shift);
      sh_need   = need - below;
    }
    __syncthreads();
    prefix = sh_prefix;
    need   = sh_need;
  }

  const unsigned thr = prefix;   // exact key of the K-th largest element
  for (int i = tid; i < DSAE; i += 256) {
    const float v = row[i];
    const unsigned key = fkey(v);
    row[i] = (key >= thr && v > 0.f) ? v : 0.f;
  }
}

// ---------------------------------------------------------------------------
extern "C" void kernel_launch(void* const* d_in, const int* in_sizes, int n_in,
                              void* d_out, int out_size, void* d_ws, size_t ws_size,
                              hipStream_t stream) {
  (void)n_in; (void)out_size; (void)d_ws; (void)ws_size;
  const float* h_seq = (const float*)d_in[0];   // (B, S, 768)
  const float* enc_w = (const float*)d_in[1];   // (S, 24576, 768)
  const float* enc_b = (const float*)d_in[2];   // (S, 24576)
  const float* dec_w = (const float*)d_in[3];   // (S, 768, 24576)
  const float* dec_b = (const float*)d_in[4];   // (S, 768)
  const int*   kptr  = (const int*)d_in[5];     // scalar k (device)

  const int B = in_sizes[0] / (S_STEPS * DMODEL);   // 1024

  float* out  = (float*)d_out;
  float* hhat = out;                                     // (B, S, 768)
  float* zreg = out + (size_t)B * S_STEPS * DMODEL;      // (B, S, 24576), also pre scratch

  dim3 blk(256);

  // 1) Encoder GEMM: pre[b,s,f] = sum_d h[b,s,d] * enc_w[s,f,d] + enc_b[s,f]
  dim3 g1(DSAE / 128, B / 128, S_STEPS);
  sae_gemm_nt_bf16<<<g1, blk, 0, stream>>>(
      h_seq, (long)DMODEL, (long)S_STEPS * DMODEL,
      enc_w, (long)DSAE * DMODEL, (long)DMODEL,
      enc_b, (long)DSAE,
      zreg,  (long)DSAE, (long)S_STEPS * DSAE,
      DMODEL);

  // 2) Top-k select + relu, in place (pre -> z)
  sae_topk_inplace<<<dim3(B * S_STEPS), blk, 0, stream>>>(zreg, kptr);

  // 3) Decoder GEMM: h_hat[b,s,d] = sum_f z[b,s,f] * dec_w[s,d,f] + dec_b[s,d]
  dim3 g3(DMODEL / 128, B / 128, S_STEPS);
  sae_gemm_nt_bf16<<<g3, blk, 0, stream>>>(
      zreg,  (long)DSAE, (long)S_STEPS * DSAE,
      dec_w, (long)DMODEL * DSAE, (long)DSAE,
      dec_b, (long)DMODEL,
      hhat,  (long)DMODEL, (long)S_STEPS * DMODEL,
      DSAE);
}
